// QuantMPTMLP_549755813919
// MI455X (gfx1250) — matrix operand order repaired and checked by reading, not compiled, not run on hardware
//
#include <hip/hip_runtime.h>
#include <math.h>
#include <stdint.h>

typedef __attribute__((ext_vector_type(16))) _Float16 v16h;
typedef __attribute__((ext_vector_type(8)))  _Float16 v8h;
typedef __attribute__((ext_vector_type(4)))  _Float16 v4h;
typedef __attribute__((ext_vector_type(2)))  _Float16 v2h;
typedef __attribute__((ext_vector_type(8)))  float    v8f;
typedef int v4i_vs __attribute__((vector_size(4 * sizeof(int))));  // matches async builtin param

#define TILE_M 128
#define TILE_N 256
#define TILE_K 32
#define LDS_STRIDE 40   // 32 data halves + 8 pad => 80B row stride (20 banks), conflict-free b128 reads

union Frag16 { v16h v; v8h h[2]; };

__device__ __forceinline__ float gelu_exact(float v) {
    return 0.5f * v * (1.0f + erff(v * 0.70710678118654752f));
}

// Async global(16B) -> LDS copy on the CDNA5 ASYNCcnt path.
__device__ __forceinline__ void async_ld16(const void* g, void* l) {
#if __has_builtin(__builtin_amdgcn_global_load_async_to_lds_b128)
    __builtin_amdgcn_global_load_async_to_lds_b128(
        (__attribute__((address_space(1))) v4i_vs*)(uintptr_t)g,
        (__attribute__((address_space(3))) v4i_vs*)(uint32_t)(uintptr_t)l,
        0, 0);
#else
    asm volatile("global_load_async_to_lds_b128 %0, %1, off"
                 :: "v"((uint32_t)(uintptr_t)l), "v"((uint64_t)(uintptr_t)g)
                 : "memory");
#endif
}

__device__ __forceinline__ void wait_async0() {
#if __has_builtin(__builtin_amdgcn_s_wait_asynccnt)
    __builtin_amdgcn_s_wait_asynccnt(0);
#else
    asm volatile("s_wait_asynccnt 0" ::: "memory");
#endif
}

// C[M,N] = act(A[M,K] @ dequant(W[K,N])), software-pipelined, double-buffered LDS.
// A_IS_F32: A is f32 (else f16, copied via async-to-LDS). APPLY_GELU: C is f16 w/ gelu (else f32).
// qweight: int32 [K, N/8] (sequential nibbles along N), qzeros: int32 [K/128, N/8],
// scales: f32 [K/128, N]. W = (q - z) * scale.
template<bool APPLY_GELU, bool A_IS_F32>
__global__ __launch_bounds__(256)
void awq_wmma_gemm(const void* __restrict__ Aptr,
                   const int*  __restrict__ qweight,
                   const int*  __restrict__ qzeros,
                   const float* __restrict__ scales,
                   void* __restrict__ Cptr,
                   int N, int K)
{
    __shared__ _Float16 ldsA[2][TILE_M * LDS_STRIDE];
    __shared__ _Float16 ldsB[2][TILE_N * LDS_STRIDE];

    const int tid      = threadIdx.x;
    const int lane     = tid & 31;
    const int wave     = tid >> 5;     // 0..7
    const int waveM    = wave >> 2;    // 0..1 : which 64-row slab
    const int waveN    = wave & 3;     // 0..3 : which 64-col slab
    const int laneHalf = lane >> 4;    // 0 = lanes 0-15, 1 = lanes 16-31
    const int lane15   = lane & 15;

    const int blockN0 = blockIdx.x * TILE_N;
    const int blockM0 = blockIdx.y * TILE_M;

    v8f acc[4][4] = {};                // 16 x (16x16 f32) accumulators per wave (64x64 tile)

    const int Ndw = N >> 3;            // dwords per qweight row

    // ---- per-thread invariants for the B (weight) pipeline ----
    const int no      = tid & 31;              // N octet (same for both tasks of a tile)
    const int kpBase  = tid >> 5;              // 0..7 ; task1 uses kpBase+8
    const int col_dw  = (blockN0 >> 3) + no;
    const float* srow = scales + blockN0 + 8 * no;

    float sv[8];                       // group-resident scales (refresh every 4 K-tiles)
    int   zw = 0;                      // group-resident zeros word
    int   q[4];                        // per-tile qweight prefetch (2 K-pairs x 2 tasks)
    float4 af[4];                      // per-tile x prefetch (f32 A path only)

    auto fetchScales = [&](int g) {
        zw = qzeros[(size_t)g * Ndw + col_dw];
        float4 s0 = *(const float4*)(srow + (size_t)g * N);
        float4 s1 = *(const float4*)(srow + (size_t)g * N + 4);
        sv[0]=s0.x; sv[1]=s0.y; sv[2]=s0.z; sv[3]=s0.w;
        sv[4]=s1.x; sv[5]=s1.y; sv[6]=s1.z; sv[7]=s1.w;
    };
    auto fetchQ = [&](int k0) {
        q[0] = qweight[(size_t)(k0 +      2 * kpBase)     * Ndw + col_dw];
        q[1] = qweight[(size_t)(k0 +      2 * kpBase + 1) * Ndw + col_dw];
        q[2] = qweight[(size_t)(k0 + 16 + 2 * kpBase)     * Ndw + col_dw];
        q[3] = qweight[(size_t)(k0 + 16 + 2 * kpBase + 1) * Ndw + col_dw];
    };
    auto storeB = [&](int buf) {       // dequant prefetched q -> [256 N][32 K] halves
        #pragma unroll
        for (int it = 0; it < 2; ++it) {
            const int kp = kpBase + 8 * it;
            const int q0 = q[2 * it], q1 = q[2 * it + 1];
            #pragma unroll
            for (int j = 0; j < 8; ++j) {
                int w0 = (q0 >> (4 * j)) & 0xF;
                int w1 = (q1 >> (4 * j)) & 0xF;
                int z  = (zw >> (4 * j)) & 0xF;
                v2h p;
                p[0] = (_Float16)((float)(w0 - z) * sv[j]);
                p[1] = (_Float16)((float)(w1 - z) * sv[j]);
                *(v2h*)&ldsB[buf][(8 * no + j) * LDS_STRIDE + 2 * kp] = p;  // 4B ds store
            }
        }
    };
    auto fetchA = [&](int k0) {        // f32 path: prefetch 4x float4 of x
        const float* A = (const float*)Aptr;
        #pragma unroll
        for (int it = 0; it < 4; ++it) {
            int task = tid + 256 * it;            // 0..1023
            int row  = task >> 3;
            int kq   = (task & 7) << 2;
            af[it] = *(const float4*)(A + (size_t)(blockM0 + row) * K + k0 + kq);
        }
    };
    auto storeA = [&](int buf) {       // f32 path: cvt + 8B ds stores
        #pragma unroll
        for (int it = 0; it < 4; ++it) {
            int task = tid + 256 * it;
            int row  = task >> 3;
            int kq   = (task & 7) << 2;
            v4h p;
            p[0] = (_Float16)af[it].x; p[1] = (_Float16)af[it].y;
            p[2] = (_Float16)af[it].z; p[3] = (_Float16)af[it].w;
            *(v4h*)&ldsA[buf][row * LDS_STRIDE + kq] = p;
        }
    };
    auto asyncA = [&](int k0, int buf) {  // f16 path: async global->LDS (16B x2)
        const _Float16* A = (const _Float16*)Aptr;
        #pragma unroll
        for (int it = 0; it < 2; ++it) {
            int task = tid + 256 * it;            // 0..511
            int row  = task >> 2;
            int kq   = (task & 3) << 3;
            async_ld16(A + (size_t)(blockM0 + row) * K + k0 + kq,
                       &ldsA[buf][row * LDS_STRIDE + kq]);
        }
    };

    // ---------------- prologue: stage tile 0 into buffer 0 ----------------
    fetchScales(0);
    fetchQ(0);
    if constexpr (A_IS_F32) { fetchA(0); storeA(0); } else { asyncA(0, 0); }
    storeB(0);

    const int nT = K >> 5;             // K / TILE_K
    for (int t = 0; t < nT; ++t) {
        if constexpr (!A_IS_F32) wait_async0();  // this wave's async LDS writes landed
        __syncthreads();                         // tile t visible; buffer ^1 free to refill

        const int cur  = t & 1;
        const int nxt  = cur ^ 1;
        const bool more = (t + 1) < nT;

        // ---- issue global fetches for tile t+1 (latency hidden under compute) ----
        if (more) {
            const int k1 = (t + 1) << 5;
            if ((k1 & 127) == 0) fetchScales(k1 >> 7);  // new quant group
            fetchQ(k1);
            if constexpr (A_IS_F32) fetchA(k1); else asyncA(k1, nxt);
        }

        // ---------------- compute tile t: 16 WMMA per wave ----------------
        // A frag (16x32 f16): lane<16 holds M=lane15, K {0..7,16..23};
        //                     lane>=16 holds M=lane15, K {8..15,24..31}.
        // B frag (32x16 f16): lane<16 holds N=lane15, K {0..7,16..23}; lane>=16 K {8..15,24..31}.
        const _Float16* aBase = &ldsA[cur][(waveM * 64 + lane15) * LDS_STRIDE + laneHalf * 8];
        const _Float16* bBase = &ldsB[cur][(waveN * 64 + lane15) * LDS_STRIDE + laneHalf * 8];

        Frag16 bfrag[4];
        #pragma unroll
        for (int ni = 0; ni < 4; ++ni) {
            const _Float16* p = bBase + ni * 16 * LDS_STRIDE;
            bfrag[ni].h[0] = *(const v8h*)(p);
            bfrag[ni].h[1] = *(const v8h*)(p + 16);
        }
        #pragma unroll
        for (int mi = 0; mi < 4; ++mi) {
            Frag16 afrag;
            const _Float16* p = aBase + mi * 16 * LDS_STRIDE;
            afrag.h[0] = *(const v8h*)(p);
            afrag.h[1] = *(const v8h*)(p + 16);
            #pragma unroll
            for (int ni = 0; ni < 4; ++ni) {
                acc[mi][ni] = __builtin_amdgcn_wmma_f32_16x16x32_f16(
                    false, afrag.v, false, bfrag[ni].v,
                    (short)0, acc[mi][ni], false, false);
            }
        }

        // ---- dequant/convert tile t+1 into buffer ^1 (data has arrived by now) ----
        if (more) {
            if constexpr (A_IS_F32) storeA(nxt);
            storeB(nxt);
        }
    }

    // ---------------- epilogue ----------------
    // C/D layout: VGPR r -> lanes 0-15: M=r, N=lane; lanes 16-31: M=r+8, N=lane-16.
    const int cRow0 = blockM0 + waveM * 64 + laneHalf * 8;
    const int cCol0 = blockN0 + waveN * 64 + lane15;
    if constexpr (APPLY_GELU) {
        _Float16* C = (_Float16*)Cptr;
        #pragma unroll
        for (int mi = 0; mi < 4; ++mi)
            #pragma unroll
            for (int ni = 0; ni < 4; ++ni)
                #pragma unroll
                for (int r = 0; r < 8; ++r) {
                    float v = gelu_exact(acc[mi][ni][r]);
                    C[(size_t)(cRow0 + mi * 16 + r) * N + cCol0 + ni * 16] = (_Float16)v;
                }
    } else {
        float* C = (float*)Cptr;
        #pragma unroll
        for (int mi = 0; mi < 4; ++mi)
            #pragma unroll
            for (int ni = 0; ni < 4; ++ni)
                #pragma unroll
                for (int r = 0; r < 8; ++r)
                    C[(size_t)(cRow0 + mi * 16 + r) * N + cCol0 + ni * 16] = acc[mi][ni][r];
    }
}

extern "C" void kernel_launch(void* const* d_in, const int* in_sizes, int n_in,
                              void* d_out, int out_size, void* d_ws, size_t ws_size,
                              hipStream_t stream) {
    (void)n_in; (void)out_size; (void)ws_size;

    const float* x      = (const float*)d_in[0];   // [B,S,D] f32
    const int*   up_qw  = (const int*)d_in[1];     // [D, F/8]
    const int*   up_qz  = (const int*)d_in[2];     // [D/128, F/8]
    const float* up_sc  = (const float*)d_in[3];   // [D/128, F]
    const int*   dn_qw  = (const int*)d_in[4];     // [F, D/8]
    const int*   dn_qz  = (const int*)d_in[5];     // [F/128, D/8]
    const float* dn_sc  = (const float*)d_in[6];   // [F/128, D]
    float*       out    = (float*)d_out;           // [T, D] f32

    const int D = 4096, F = 16384;
    const int T = in_sizes[0] / D;                 // 8192 tokens

    _Float16* h = (_Float16*)d_ws;                 // [T, F] f16 intermediate (256 MB)

    dim3 blk(256);
    dim3 g1(F / TILE_N, T / TILE_M);               // 64 x 64 blocks
    awq_wmma_gemm<true,  true ><<<g1, blk, 0, stream>>>(x, up_qw, up_qz, up_sc, (void*)h, F, D);

    dim3 g2(D / TILE_N, T / TILE_M);               // 16 x 64 blocks
    awq_wmma_gemm<false, false><<<g2, blk, 0, stream>>>((const void*)h, dn_qw, dn_qz, dn_sc, (void*)out, D, F);
}
